// NeuralPhysicsPredictor_47528108097594
// MI455X (gfx1250) — matrix operand order repaired and checked by reading, not compiled, not run on hardware
//
#include <hip/hip_runtime.h>

// ---------------------------------------------------------------------------
// Types for WMMA operands (gfx1250, wave32)
// ---------------------------------------------------------------------------
typedef __bf16 bf16_t;
typedef bf16_t v16bf __attribute__((ext_vector_type(16)));
typedef float  v8f   __attribute__((ext_vector_type(8)));

struct alignas(16) U4x { unsigned int x, y, z, w; };
struct alignas(16) F4x { float x, y, z, w; };

union V16 { v16bf v; unsigned short s[16]; U4x q[2]; };
union V8  { v8f   v; float f[8]; };

__device__ __forceinline__ unsigned short f2bf(float f) {
  unsigned int u = __builtin_bit_cast(unsigned int, f);
  unsigned int r = u + 0x7FFFu + ((u >> 16) & 1u);   // round-to-nearest-even
  return (unsigned short)(r >> 16);
}

// ---------------------------------------------------------------------------
// Weight convert + transpose: Wt[n*Kp + k] = bf16(W[k*N + n]), zero-pad k>=K
// ---------------------------------------------------------------------------
__global__ void cvt_transpose(const float* __restrict__ W,
                              unsigned short* __restrict__ Wt,
                              int K, int N, int Kp) {
  int idx = blockIdx.x * 256 + threadIdx.x;
  if (idx >= N * Kp) return;
  int n = idx / Kp, k = idx - n * Kp;
  float v = (k < K) ? W[(size_t)k * N + n] : 0.f;
  Wt[idx] = f2bf(v);
}

// object_states [1600,19] -> bf16 [1600,32] zero-padded
__global__ void build_x0(const float* __restrict__ obj,
                         unsigned short* __restrict__ X0) {
  int idx = blockIdx.x * 256 + threadIdx.x;
  if (idx >= 1600 * 32) return;
  int r = idx >> 5, k = idx & 31;
  X0[idx] = f2bf(k < 19 ? obj[r * 19 + k] : 0.f);
}

__global__ void zero_f32(float* __restrict__ p, int n) {
  int i = blockIdx.x * 256 + threadIdx.x;
  if (i < n) p[i] = 0.f;
}

// ---------------------------------------------------------------------------
// Generic bf16 WMMA GEMM: out = act(A[M,Kp] @ Wt[N,Kp]^T + bias + residual)
// block = 64 threads (2 waves); each wave: 16(M) x 64(N) via 4 accumulators.
// grid = (N/128, M/16). A row-major bf16, Wt transposed bf16.
// ---------------------------------------------------------------------------
__global__ void gemm_bf16_wmma(const unsigned short* __restrict__ A,
                               const unsigned short* __restrict__ Wt,
                               const float* __restrict__ bias,
                               const float* __restrict__ residual,
                               float* __restrict__ outF,
                               unsigned short* __restrict__ outB,
                               int Kp, int N, int relu) {
  const int lane = threadIdx.x & 31;
  const int wid  = threadIdx.x >> 5;
  const int mr   = lane & 15;
  const int half = lane >> 4;
  const int m0 = blockIdx.y * 16;
  const int n0 = blockIdx.x * 128 + wid * 64;

  V8 acc[4];
#pragma unroll
  for (int t = 0; t < 4; ++t)
#pragma unroll
    for (int r = 0; r < 8; ++r) acc[t].f[r] = 0.f;

  const unsigned short* arow = A + (size_t)(m0 + mr) * Kp;
  for (int k0 = 0; k0 < Kp; k0 += 32) {
    V16 va;
    va.q[0] = *(const U4x*)(arow + k0 + half * 8);        // elems 0..7 : K=h*8+e
    va.q[1] = *(const U4x*)(arow + k0 + 16 + half * 8);   // elems 8..15: K=16+h*8+e
#pragma unroll
    for (int t = 0; t < 4; ++t) {
      const unsigned short* brow = Wt + (size_t)(n0 + t * 16 + mr) * Kp;
      V16 vb;
      vb.q[0] = *(const U4x*)(brow + k0 + half * 8);      // K = h*8 + e
      vb.q[1] = *(const U4x*)(brow + k0 + 16 + half * 8);
      acc[t].v = __builtin_amdgcn_wmma_f32_16x16x32_bf16(
          false, va.v, false, vb.v, (short)0, acc[t].v, false, false);
    }
  }

#pragma unroll
  for (int t = 0; t < 4; ++t) {
    int n = n0 + t * 16 + mr;
    float bv = bias ? bias[n] : 0.f;
#pragma unroll
    for (int r = 0; r < 8; ++r) {
      int m = m0 + r + 8 * half;
      float v = acc[t].f[r] + bv;
      if (residual) v += residual[(size_t)m * N + n];
      if (relu) v = fmaxf(v, 0.f);
      size_t o = (size_t)m * N + n;
      if (outF) outF[o] = v;
      if (outB) outB[o] = f2bf(v);
    }
  }
}

// ---------------------------------------------------------------------------
// Attention: one block per (b,h); thread i = query; online softmax (flash).
// ---------------------------------------------------------------------------
__global__ void attention_kernel(const float* __restrict__ qf,
                                 const float* __restrict__ kf,
                                 const float* __restrict__ vf,
                                 float* __restrict__ ctxF,
                                 unsigned short* __restrict__ ctxB) {
  int bh = blockIdx.x;
  int b = bh >> 3, h = bh & 7;
  int i = threadIdx.x;
  if (i >= 100) return;
  size_t base = ((size_t)(b * 100 + i)) * 512 + h * 64;
  float qv[64];
  const float* qr = qf + base;
#pragma unroll
  for (int d = 0; d < 64; ++d) qv[d] = qr[d];
  float mx = -3.0e38f, l = 0.f;
  float acc[64];
#pragma unroll
  for (int d = 0; d < 64; ++d) acc[d] = 0.f;
  for (int j = 0; j < 100; ++j) {
    size_t kb = ((size_t)(b * 100 + j)) * 512 + h * 64;
    const float* kr = kf + kb;
    float s = 0.f;
#pragma unroll 16
    for (int d = 0; d < 64; ++d) s += qv[d] * kr[d];
    s *= 0.125f;                       // 1/sqrt(64)
    float nm = fmaxf(mx, s);
    float cs = __expf(mx - nm);
    float p  = __expf(s - nm);
    l = l * cs + p;
    const float* vr = vf + kb;
#pragma unroll 16
    for (int d = 0; d < 64; ++d) acc[d] = acc[d] * cs + p * vr[d];
    mx = nm;
  }
  float inv = 1.f / l;
  float* of = ctxF + base;
  unsigned short* ob = ctxB + base;
#pragma unroll 16
  for (int d = 0; d < 64; ++d) {
    float v = acc[d] * inv;
    of[d] = v;
    ob[d] = f2bf(v);
  }
}

// ---------------------------------------------------------------------------
// Dynamics final layer 256->13, fused *dt and gravity on dims 3:6
// ---------------------------------------------------------------------------
__global__ void dyn_final(const float* __restrict__ h, const float* __restrict__ W3,
                          const float* __restrict__ b3, const float* __restrict__ dtp,
                          const float* __restrict__ gfp, float* __restrict__ out) {
  int row = blockIdx.x;
  int o = threadIdx.x;
  if (o >= 13) return;
  const float* hr = h + (size_t)row * 256;
  float s = 0.f;
#pragma unroll 8
  for (int k = 0; k < 256; ++k) s += hr[k] * W3[k * 13 + o];
  float dt = dtp[0];
  float v = (s + b3[o]) * dt;
  if (o >= 3 && o < 6) v += gfp[o - 3] * dt;
  out[(size_t)row * 13 + o] = v;
}

// ---------------------------------------------------------------------------
// Uncertainty head: 512 -> 128 relu -> 1 sigmoid (block per row)
// ---------------------------------------------------------------------------
__global__ void ue_kernel(const float* __restrict__ e, const float* __restrict__ W1,
                          const float* __restrict__ b1, const float* __restrict__ W2,
                          const float* __restrict__ b2, float* __restrict__ out) {
  __shared__ float red[128];
  int row = blockIdx.x, t = threadIdx.x;
  const float* er = e + (size_t)row * 512;
  float s = 0.f;
#pragma unroll 8
  for (int k = 0; k < 512; ++k) s += er[k] * W1[k * 128 + t];
  s = fmaxf(s + b1[t], 0.f);
  red[t] = s * W2[t];
  __syncthreads();
  for (int stp = 64; stp > 0; stp >>= 1) {
    if (t < stp) red[t] += red[t + stp];
    __syncthreads();
  }
  if (t == 0) out[row] = 1.f / (1.f + __expf(-(red[0] + b2[0])));
}

// ---------------------------------------------------------------------------
// Fused contact: h1 = relu(Ap[i] + Bp[j]) built in registers -> WMMA with
// cd_W2 (256 hidden) -> relu -> 256->4 projection via shfl reduce -> write
// (b,i,j) and (b,j,i) for i<j. Block = 128 threads (4 waves x 64 hidden).
// grid = (7 j-tiles, 100 i, 16 b); tiles entirely j<=i exit early.
// ---------------------------------------------------------------------------
__global__ void contact_fused(const float* __restrict__ Ap,
                              const float* __restrict__ Bp,
                              const unsigned short* __restrict__ Wt2,  // [256][512]
                              const float* __restrict__ b2,
                              const float* __restrict__ W3,            // [256,4]
                              const float* __restrict__ b3,
                              float* __restrict__ out) {               // [16,100,100,4]
  const int b = blockIdx.z;
  const int i = blockIdx.y;
  const int j0 = blockIdx.x * 16;
  int jmax = j0 + 15; if (jmax > 99) jmax = 99;
  if (jmax <= i) return;   // uniform exit: no j>i in this tile

  __shared__ float part[16][4];
  if (threadIdx.x < 64) part[threadIdx.x >> 2][threadIdx.x & 3] = 0.f;
  __syncthreads();

  const int lane = threadIdx.x & 31;
  const int wid  = threadIdx.x >> 5;
  const int mr = lane & 15, half = lane >> 4;
  const int j = j0 + mr;
  const int jc = j < 100 ? j : 99;           // clamp; stores are masked later
  const float* arow = Ap + (size_t)(b * 100 + i) * 512;
  const float* brow = Bp + (size_t)(b * 100 + jc) * 512;
  const int n0 = wid * 64;

  V8 acc[4];
#pragma unroll
  for (int t = 0; t < 4; ++t)
#pragma unroll
    for (int r = 0; r < 8; ++r) acc[t].f[r] = 0.f;

  for (int k0 = 0; k0 < 512; k0 += 32) {
    V16 va;
#pragma unroll
    for (int c = 0; c < 2; ++c) {
      const int base = k0 + half * 8 + c * 16;
      F4x a0 = *(const F4x*)(arow + base);
      F4x a1 = *(const F4x*)(arow + base + 4);
      F4x p0 = *(const F4x*)(brow + base);
      F4x p1 = *(const F4x*)(brow + base + 4);
      va.s[c * 8 + 0] = f2bf(fmaxf(a0.x + p0.x, 0.f));
      va.s[c * 8 + 1] = f2bf(fmaxf(a0.y + p0.y, 0.f));
      va.s[c * 8 + 2] = f2bf(fmaxf(a0.z + p0.z, 0.f));
      va.s[c * 8 + 3] = f2bf(fmaxf(a0.w + p0.w, 0.f));
      va.s[c * 8 + 4] = f2bf(fmaxf(a1.x + p1.x, 0.f));
      va.s[c * 8 + 5] = f2bf(fmaxf(a1.y + p1.y, 0.f));
      va.s[c * 8 + 6] = f2bf(fmaxf(a1.z + p1.z, 0.f));
      va.s[c * 8 + 7] = f2bf(fmaxf(a1.w + p1.w, 0.f));
    }
#pragma unroll
    for (int t = 0; t < 4; ++t) {
      const unsigned short* wrow = Wt2 + (size_t)(n0 + t * 16 + mr) * 512;
      V16 vb;
      vb.q[0] = *(const U4x*)(wrow + k0 + half * 8);
      vb.q[1] = *(const U4x*)(wrow + k0 + 16 + half * 8);
      acc[t].v = __builtin_amdgcn_wmma_f32_16x16x32_bf16(
          false, va.v, false, vb.v, (short)0, acc[t].v, false, false);
    }
  }

  // Layer 3: h2 (per-lane slices) -> 4 outputs, reduce across the 16 lanes
  // of each half-wave (m = r + 8*half), then across waves via LDS atomics.
#pragma unroll
  for (int r = 0; r < 8; ++r) {
    float s0 = 0.f, s1 = 0.f, s2 = 0.f, s3 = 0.f;
#pragma unroll
    for (int t = 0; t < 4; ++t) {
      int n = n0 + t * 16 + mr;
      float hv = fmaxf(acc[t].f[r] + b2[n], 0.f);
      s0 += hv * W3[n * 4 + 0];
      s1 += hv * W3[n * 4 + 1];
      s2 += hv * W3[n * 4 + 2];
      s3 += hv * W3[n * 4 + 3];
    }
#pragma unroll
    for (int off = 1; off < 16; off <<= 1) {
      s0 += __shfl_xor(s0, off, 32);
      s1 += __shfl_xor(s1, off, 32);
      s2 += __shfl_xor(s2, off, 32);
      s3 += __shfl_xor(s3, off, 32);
    }
    if (mr == 0) {
      int m = r + 8 * half;
      atomicAdd(&part[m][0], s0);
      atomicAdd(&part[m][1], s1);
      atomicAdd(&part[m][2], s2);
      atomicAdd(&part[m][3], s3);
    }
  }
  __syncthreads();
  if (threadIdx.x < 64) {
    int m = threadIdx.x >> 2, o = threadIdx.x & 3;
    int jj = j0 + m;
    if (jj < 100 && jj > i) {
      float v = part[m][o] + b3[o];
      out[(((size_t)b * 100 + i) * 100 + jj) * 4 + o] = v;   // (i,j)
      out[(((size_t)b * 100 + jj) * 100 + i) * 4 + o] = v;   // mirror (j,i)
    }
  }
}

// ---------------------------------------------------------------------------
// Host orchestration
// ---------------------------------------------------------------------------
extern "C" void kernel_launch(void* const* d_in, const int* in_sizes, int n_in,
                              void* d_out, int out_size, void* d_ws, size_t ws_size,
                              hipStream_t stream) {
  const float* obj  = (const float*)d_in[0];
  const float* dtp  = (const float*)d_in[1];
  const float* gfp  = (const float*)d_in[2];
  const float* seW1 = (const float*)d_in[3];  const float* seb1 = (const float*)d_in[4];
  const float* seW2 = (const float*)d_in[5];  const float* seb2 = (const float*)d_in[6];
  const float* seW3 = (const float*)d_in[7];  const float* seb3 = (const float*)d_in[8];
  const float* Wq   = (const float*)d_in[9];  const float* bq   = (const float*)d_in[10];
  const float* Wk   = (const float*)d_in[11]; const float* bk   = (const float*)d_in[12];
  const float* Wv   = (const float*)d_in[13]; const float* bv   = (const float*)d_in[14];
  const float* Wo   = (const float*)d_in[15]; const float* bo   = (const float*)d_in[16];
  const float* dpW1 = (const float*)d_in[17]; const float* dpb1 = (const float*)d_in[18];
  const float* dpW2 = (const float*)d_in[19]; const float* dpb2 = (const float*)d_in[20];
  const float* dpW3 = (const float*)d_in[21]; const float* dpb3 = (const float*)d_in[22];
  const float* cdW1 = (const float*)d_in[23]; const float* cdb1 = (const float*)d_in[24];
  const float* cdW2 = (const float*)d_in[25]; const float* cdb2 = (const float*)d_in[26];
  const float* cdW3 = (const float*)d_in[27]; const float* cdb3 = (const float*)d_in[28];
  const float* ueW1 = (const float*)d_in[29]; const float* ueb1 = (const float*)d_in[30];
  const float* ueW2 = (const float*)d_in[31]; const float* ueb2 = (const float*)d_in[32];

  constexpr int M = 1600;   // B*N rows
  char* wsb = (char*)d_ws;
  size_t off = 0;
  auto alloc = [&](size_t bytes) -> void* {
    off = (off + 255) & ~(size_t)255;
    void* p = (void*)(wsb + off);
    off += bytes;
    return p;
  };
  auto aU = [&](size_t n) { return (unsigned short*)alloc(n * 2); };
  auto aF = [&](size_t n) { return (float*)alloc(n * 4); };

  // transposed bf16 weights
  unsigned short* wt_se1  = aU(512 * 32);
  unsigned short* wt_se2  = aU(512 * 512);
  unsigned short* wt_se3  = aU(512 * 512);
  unsigned short* wt_q    = aU(512 * 512);
  unsigned short* wt_k    = aU(512 * 512);
  unsigned short* wt_v    = aU(512 * 512);
  unsigned short* wt_o    = aU(512 * 512);
  unsigned short* wt_dp1  = aU(512 * 512);
  unsigned short* wt_dp2  = aU(256 * 512);
  unsigned short* wt_cd1a = aU(512 * 512);
  unsigned short* wt_cd1b = aU(512 * 512);
  unsigned short* wt_cd2  = aU(256 * 512);
  // activations
  unsigned short* x0b  = aU((size_t)M * 32);
  unsigned short* e1b  = aU((size_t)M * 512);
  unsigned short* e2b  = aU((size_t)M * 512);
  unsigned short* eb   = aU((size_t)M * 512);
  unsigned short* ctxb = aU((size_t)M * 512);
  unsigned short* epb  = aU((size_t)M * 512);
  unsigned short* d1b  = aU((size_t)M * 512);
  float* ef   = aF((size_t)M * 512);
  float* qf   = aF((size_t)M * 512);
  float* kf   = aF((size_t)M * 512);
  float* vf   = aF((size_t)M * 512);
  float* ctxf = aF((size_t)M * 512);
  float* epf  = aF((size_t)M * 512);
  float* d2f  = aF((size_t)M * 256);
  float* apf  = aF((size_t)M * 512);
  float* bpf  = aF((size_t)M * 512);

  float* outp        = (float*)d_out;
  float* out_sc      = outp;                       // [16,100,13]
  float* out_contact = outp + 20800;               // [16,100,100,4]
  float* out_u       = outp + 20800 + 640000;      // [16,100,1]

  auto cvt = [&](const float* W, unsigned short* Wt, int K, int N, int Kp) {
    int n = N * Kp;
    cvt_transpose<<<dim3((n + 255) / 256), dim3(256), 0, stream>>>(W, Wt, K, N, Kp);
  };
  cvt(seW1, wt_se1, 19, 512, 32);
  cvt(seW2, wt_se2, 512, 512, 512);
  cvt(seW3, wt_se3, 512, 512, 512);
  cvt(Wq, wt_q, 512, 512, 512);
  cvt(Wk, wt_k, 512, 512, 512);
  cvt(Wv, wt_v, 512, 512, 512);
  cvt(Wo, wt_o, 512, 512, 512);
  cvt(dpW1, wt_dp1, 512, 512, 512);
  cvt(dpW2, wt_dp2, 512, 256, 512);
  cvt(cdW1, wt_cd1a, 512, 512, 512);               // rows 0..511 of cd_W1
  cvt(cdW1 + (size_t)512 * 512, wt_cd1b, 512, 512, 512);  // rows 512..1023
  cvt(cdW2, wt_cd2, 512, 256, 512);
  build_x0<<<dim3((1600 * 32 + 255) / 256), dim3(256), 0, stream>>>(obj, x0b);

  auto gemm = [&](const unsigned short* A, const unsigned short* Wt, const float* bias,
                  const float* res, float* oF, unsigned short* oB, int Kp, int N, int relu) {
    gemm_bf16_wmma<<<dim3(N / 128, M / 16), dim3(64), 0, stream>>>(
        A, Wt, bias, res, oF, oB, Kp, N, relu);
  };

  // state encoder
  gemm(x0b, wt_se1, seb1, nullptr, nullptr, e1b, 32, 512, 1);
  gemm(e1b, wt_se2, seb2, nullptr, nullptr, e2b, 512, 512, 1);
  gemm(e2b, wt_se3, seb3, nullptr, ef, eb, 512, 512, 0);
  // attention
  gemm(eb, wt_q, bq, nullptr, qf, nullptr, 512, 512, 0);
  gemm(eb, wt_k, bk, nullptr, kf, nullptr, 512, 512, 0);
  gemm(eb, wt_v, bv, nullptr, vf, nullptr, 512, 512, 0);
  attention_kernel<<<dim3(128), dim3(128), 0, stream>>>(qf, kf, vf, ctxf, ctxb);
  gemm(ctxb, wt_o, bo, ef, epf, epb, 512, 512, 0);   // e' = e + ctx@Wo + bo
  // dynamics
  gemm(epb, wt_dp1, dpb1, nullptr, nullptr, d1b, 512, 512, 1);
  gemm(d1b, wt_dp2, dpb2, nullptr, d2f, nullptr, 512, 256, 1);
  dyn_final<<<dim3(1600), dim3(32), 0, stream>>>(d2f, dpW3, dpb3, dtp, gfp, out_sc);
  // contact: decomposed layer 1, then fused pair kernel
  gemm(epb, wt_cd1a, cdb1, nullptr, apf, nullptr, 512, 512, 0);
  gemm(epb, wt_cd1b, nullptr, nullptr, bpf, nullptr, 512, 512, 0);
  zero_f32<<<dim3((640000 + 255) / 256), dim3(256), 0, stream>>>(out_contact, 640000);
  contact_fused<<<dim3(7, 100, 16), dim3(128), 0, stream>>>(
      apf, bpf, wt_cd2, cdb2, cdW3, cdb3, out_contact);
  // uncertainty
  ue_kernel<<<dim3(1600), dim3(128), 0, stream>>>(epf, ueW1, ueb1, ueW2, ueb2, out_u);

  (void)in_sizes; (void)n_in; (void)out_size; (void)ws_size;
}
